// GAT_13821204758567
// MI455X (gfx1250) — matrix operand order repaired and checked by reading, not compiled, not run on hardware
//
#include <hip/hip_runtime.h>
#include <stdint.h>

#define NEG_SLOPE 0.2f
#define EPS_DEN   1e-16f

typedef __attribute__((ext_vector_type(8)))  float  v8f;
typedef __attribute__((ext_vector_type(16))) __bf16 bf16x16;

union BF16x16U { bf16x16 v; unsigned u[8]; };

static __device__ __forceinline__ unsigned f2bf(float f) {
  unsigned u = __float_as_uint(f);
  return (u + 0x7FFFu + ((u >> 16) & 1u)) >> 16;   // RNE fp32 -> bf16
}

// ---------------- utility / CSR build ----------------

__global__ void k_fill_i32(int* p, int v, int n) {
  int g = blockIdx.x * blockDim.x + threadIdx.x;
  if (g < n) p[g] = v;
}

__global__ void k_count_deg(const int* __restrict__ dst, int E, int N, int* deg) {
  int e = blockIdx.x * blockDim.x + threadIdx.x;
  if (e >= E + N) return;
  int d = (e < E) ? dst[e] : (e - E);          // self loop: dst = e - E
  atomicAdd(&deg[d], 1);
}

__global__ void k_scan_block(const int* __restrict__ in, int* out, int* bsums, int n) {
  __shared__ int tmp[256];
  int t = threadIdx.x;
  int g = blockIdx.x * 256 + t;
  int v = (g < n) ? in[g] : 0;
  tmp[t] = v; __syncthreads();
  for (int off = 1; off < 256; off <<= 1) {
    int x = (t >= off) ? tmp[t - off] : 0;
    __syncthreads();
    tmp[t] += x;
    __syncthreads();
  }
  if (g < n) out[g] = tmp[t] - v;              // exclusive within block
  if (t == 255) bsums[blockIdx.x] = tmp[t];
}

__global__ void k_scan_sums(int* bsums, int nb) {
  __shared__ int tmp[256];
  int t = threadIdx.x;
  int v = (t < nb) ? bsums[t] : 0;
  tmp[t] = v; __syncthreads();
  for (int off = 1; off < 256; off <<= 1) {
    int x = (t >= off) ? tmp[t - off] : 0;
    __syncthreads();
    tmp[t] += x;
    __syncthreads();
  }
  if (t < nb) bsums[t] = tmp[t] - v;           // exclusive block offsets
}

__global__ void k_add_off(int* rowptr, const int* __restrict__ bsums, int* cursor,
                          int n, int total) {
  int g = blockIdx.x * blockDim.x + threadIdx.x;
  if (g < n) {
    int r = rowptr[g] + bsums[g >> 8];
    rowptr[g] = r;
    cursor[g] = r;
  }
  if (g == 0) rowptr[n] = total;
}

__global__ void k_scatter(const int* __restrict__ dst, int E, int N,
                          int* cursor, int* eidx) {
  int e = blockIdx.x * blockDim.x + threadIdx.x;
  if (e >= E + N) return;
  int d = (e < E) ? dst[e] : (e - E);
  int pos = atomicAdd(&cursor[d], 1);
  eidx[pos] = e;
}

// ---------------- bf16 packing for WMMA ----------------

// Activations: row-major [M,K] -> packed (k,k+1) pairs, one dword per pair.
__global__ void k_pack_act(const float* __restrict__ in, unsigned* __restrict__ out,
                           int npairs) {
  int g = blockIdx.x * blockDim.x + threadIdx.x;
  if (g >= npairs) return;
  out[g] = f2bf(in[2 * g]) | (f2bf(in[2 * g + 1]) << 16);
}

// Weights: [K,Nn] row-major -> pair rows (2t,2t+1) packed per column.
__global__ void k_pack_wt(const float* __restrict__ in, unsigned* __restrict__ out,
                          int Kp, int Nn) {
  int g = blockIdx.x * blockDim.x + threadIdx.x;
  if (g >= Kp * Nn) return;
  int t = g / Nn, n = g - t * Nn;
  float lo = in[(size_t)(2 * t) * Nn + n];
  float hi = in[(size_t)(2 * t + 1) * Nn + n];
  out[g] = f2bf(lo) | (f2bf(hi) << 16);
}

// ---------------- WMMA GEMM: C[M,Nn] = A[M,K] * B[K,Nn], bf16 in / f32 out ---------------
// Block = 8 waves = 128(M) x 64(N) supertile. Per 32-deep k-step the packed B tile
// (16 rows x 64 dwords) is staged into LDS with global_load_async_to_lds_b128
// (ASYNCcnt path), then each wave runs 4 WMMAs (16x64) reusing its A fragment.
// LDS row stride 68 dwords: lanes 0-15 (K 0..7,16..23) and 16-31 (K 8..15,24..31) hit disjoint banks.
__global__ void __launch_bounds__(256)
k_wmma_gemm_bf16(const unsigned* __restrict__ apk,
                 const unsigned* __restrict__ bpk,
                 float* __restrict__ cout,
                 int M, int K, int Nn) {
  __shared__ unsigned Bt[16 * 68];
  const int tid  = threadIdx.x;
  const int wv   = tid >> 5;
  const int lane = tid & 31;
  const int nsuper = Nn >> 6;                    // N supertiles per grid row
  const int bm = blockIdx.x / nsuper;
  const int bn = blockIdx.x - bm * nsuper;
  const int l15 = lane & 15;
  const int g   = lane >> 4;                     // lane group
  const int mrow = bm * 128 + wv * 16 + l15;     // A row read by this lane (Mpad-safe)
  const int Kp = K >> 1;
  const unsigned* arow = apk + (size_t)mrow * Kp;

  // async staging coords: thread -> B-tile row r (0..15), 16B chunk q (0..15)
  const int r = tid >> 4, q = tid & 15;
  const unsigned ldsa = (unsigned)(size_t)(&Bt[r * 68 + q * 4]);
  const unsigned* gsrc = bpk + (size_t)r * Nn + bn * 64 + q * 4;

  v8f acc[4];
#pragma unroll
  for (int j = 0; j < 4; ++j) acc[j] = (v8f){0.f,0.f,0.f,0.f,0.f,0.f,0.f,0.f};

  const int ksteps = K >> 5;
  for (int kk = 0; kk < ksteps; ++kk) {
    // ---- DMA B tile to LDS (each thread: one 16B async transfer) ----
    {
      unsigned long long ga =
          (unsigned long long)(size_t)(gsrc + (size_t)kk * 16 * Nn);
      asm volatile("global_load_async_to_lds_b128 %0, %1, off"
                   :: "v"(ldsa), "v"(ga) : "memory");
      asm volatile("s_wait_asynccnt 0x0" ::: "memory");
    }
    __syncthreads();

    // ---- A fragment: V0-3 -> K 0..7 (+8g), V4-7 -> K 16..23 (+8g) ----
    BF16x16U A;
    const int ab = kk * 16 + 4 * g;
#pragma unroll
    for (int v = 0; v < 8; ++v)
      A.u[v] = arow[ab + (v < 4 ? v : v + 4)];

    // ---- 4 WMMAs over N subtiles, B from LDS ----
    const int brow = 4 * g;                       // lanes 0-15: rows 0..3,8..11; 16-31: 4..7,12..15
#pragma unroll
    for (int j = 0; j < 4; ++j) {
      BF16x16U Bm;
#pragma unroll
      for (int v = 0; v < 8; ++v)
        Bm.u[v] = Bt[(brow + (v < 4 ? v : v + 4)) * 68 + j * 16 + l15];
      acc[j] = __builtin_amdgcn_wmma_f32_16x16x32_bf16(
          false, A.v, false, Bm.v, (short)0, acc[j], false, false);
    }
    __syncthreads();                              // protect Bt before restage
  }

  // ---- epilogue: C VGPR r -> M = r + 8g, N = lane&15 ; guard pad rows ----
  const int colbase = bn * 64 + l15;
  const int rowbase = bm * 128 + wv * 16 + 8 * g;
#pragma unroll
  for (int j = 0; j < 4; ++j) {
#pragma unroll
    for (int rr = 0; rr < 8; ++rr) {
      int row = rowbase + rr;
      if (row < M) cout[(size_t)row * Nn + colbase + j * 16] = acc[j][rr];
    }
  }
}

// ---------------- per-node attention logits: es/ed = <h[n,h,:], a> ----------------
__global__ void k_node_alpha(const float* __restrict__ feat,
                             const float* __restrict__ a_src,
                             const float* __restrict__ a_dst,
                             float* es, float* ed, int NH, int H, int C) {
  int g = blockIdx.x * blockDim.x + threadIdx.x;
  if (g >= NH) return;
  int h = g % H;
  const float* f  = feat + (size_t)g * C;       // row (n*H+h) of [N, H*C]
  const float* as = a_src + h * C;
  const float* ad = a_dst + h * C;
  float s = 0.f, d = 0.f;
  for (int c = 0; c < C; ++c) {
    float v = f[c];
    s += v * as[c];
    d += v * ad[c];
  }
  es[g] = s;
  ed[g] = d;
}

// ---------------- fused segment softmax + aggregation (one-pass online softmax) ----------
// One wave per (dst node, head); lane holds channels c and c+32 (C == 64).
__global__ void k_gat_agg(const float* __restrict__ feat,
                          const float* __restrict__ es,
                          const float* __restrict__ ed,
                          const int* __restrict__ rowptr,
                          const int* __restrict__ eidx,
                          const int* __restrict__ srcArr,
                          const float* __restrict__ bias,
                          float* __restrict__ out,
                          int E, int N, int H, int C, int ldo, int relu) {
  int wave = (int)((blockIdx.x * blockDim.x + threadIdx.x) >> 5);
  int lane = threadIdx.x & 31;
  if (wave >= N * H) return;
  int n = wave / H, h = wave - n * H;
  int c0 = lane, c1 = lane + 32;
  float edn = ed[wave];
  int beg = rowptr[n], end = rowptr[n + 1];
  float m = -3.0e38f, d = 0.f, a0 = 0.f, a1 = 0.f;
  for (int i = beg; i < end; ++i) {
    __builtin_prefetch(&eidx[i + 16], 0, 0);
    int e = eidx[i];
    int s = (e < E) ? srcArr[e] : (e - E);      // self loop: src == dst node
    float sc = es[(size_t)s * H + h] + edn;
    sc = sc > 0.f ? sc : NEG_SLOPE * sc;        // leaky_relu(0.2)
    float mn    = fmaxf(m, sc);
    float scale = __expf(m - mn);
    float p     = __expf(sc - mn);
    const float* f = feat + ((size_t)s * H + h) * C;
    d  = d * scale + p;
    a0 = a0 * scale + p * f[c0];
    a1 = a1 * scale + p * f[c1];
    m = mn;
  }
  float r  = 1.f / (d + EPS_DEN);
  float o0 = a0 * r + bias[h * C + c0];
  float o1 = a1 * r + bias[h * C + c1];
  if (relu) { o0 = fmaxf(o0, 0.f); o1 = fmaxf(o1, 0.f); }
  out[(size_t)n * ldo + h * C + c0] = o0;
  out[(size_t)n * ldo + h * C + c1] = o1;
}

// ---------------- host orchestration ----------------

extern "C" void kernel_launch(void* const* d_in, const int* in_sizes, int n_in,
                              void* d_out, int out_size, void* d_ws, size_t ws_size,
                              hipStream_t stream) {
  (void)in_sizes; (void)n_in; (void)out_size; (void)ws_size;
  const int Bn = 2, N = 20000, E = 320000, Fin = 128, H = 8, C = 64;
  const int HC = H * C;                 // 512
  const int Etot = E + N;               // edges + self loops
  const int Mpad = ((N + 127) / 128) * 128;   // 20096: GEMM supertile padding

  const float* xs     = (const float*)d_in[0];
  const int*   eis    = (const int*)  d_in[1];
  const float* W1     = (const float*)d_in[2];
  const float* a_src1 = (const float*)d_in[3];
  const float* a_dst1 = (const float*)d_in[4];
  const float* b1     = (const float*)d_in[5];
  const float* W2     = (const float*)d_in[6];
  const float* a_src2 = (const float*)d_in[7];
  const float* a_dst2 = (const float*)d_in[8];
  const float* b2     = (const float*)d_in[9];
  float* outp = (float*)d_out;

  char* w = (char*)d_ws;
  size_t off = 0;
  auto alloc = [&](size_t bytes) -> char* {
    char* p = w + off;
    off = (off + bytes + 255) & ~(size_t)255;
    return p;
  };
  float*    h1     = (float*)   alloc((size_t)N * HC * 4);
  float*    hp     = (float*)   alloc((size_t)N * HC * 4);
  float*    h2     = (float*)   alloc((size_t)N * C * 4);
  float*    es1    = (float*)   alloc((size_t)N * H * 4);
  float*    ed1    = (float*)   alloc((size_t)N * H * 4);
  float*    es2    = (float*)   alloc((size_t)N * 4);
  float*    ed2    = (float*)   alloc((size_t)N * 4);
  unsigned* apk    = (unsigned*)alloc((size_t)Mpad * HC * 2); // padded; reused x / hp
  unsigned* w1pk   = (unsigned*)alloc((size_t)Fin * HC * 2);
  unsigned* w2pk   = (unsigned*)alloc((size_t)HC * C * 2);
  int*      rowptr = (int*)     alloc((size_t)(N + 1) * 4);
  int*      cursor = (int*)     alloc((size_t)N * 4);
  int*      bsums  = (int*)     alloc(256 * 4);
  int*      eidx   = (int*)     alloc((size_t)Etot * 4);

  const int TB = 256;
  auto cdiv = [](long long a, long long b) { return (int)((a + b - 1) / b); };

  // pack weights to bf16 once (shared by both batches)
  k_pack_wt<<<cdiv((long long)(Fin / 2) * HC, TB), TB, 0, stream>>>(W1, w1pk, Fin / 2, HC);
  k_pack_wt<<<cdiv((long long)(HC / 2) * C,  TB), TB, 0, stream>>>(W2, w2pk, HC / 2, C);

  const int nbScan = cdiv(N, 256);      // 79 blocks -> single-block second-level scan

  for (int b = 0; b < Bn; ++b) {
    const float* x   = xs + (size_t)b * N * Fin;
    const int*   src = eis + (size_t)b * 2 * E;
    const int*   dst = src + E;
    float*       ob  = outp + (size_t)b * N * C;

    // ---- CSR (dst-sorted edge ids), shared by both layers ----
    k_fill_i32 <<<cdiv(N, TB),    TB, 0, stream>>>(cursor, 0, N);
    k_count_deg<<<cdiv(Etot, TB), TB, 0, stream>>>(dst, E, N, cursor);
    k_scan_block<<<nbScan, 256, 0, stream>>>(cursor, rowptr, bsums, N);
    k_scan_sums <<<1, 256, 0, stream>>>(bsums, nbScan);
    k_add_off   <<<nbScan, 256, 0, stream>>>(rowptr, bsums, cursor, N, Etot);
    k_scatter   <<<cdiv(Etot, TB), TB, 0, stream>>>(dst, E, N, cursor, eidx);

    // ---- layer 1: h1 = x @ W1 (WMMA + async-LDS B), logits, fused agg + bias + relu ----
    k_pack_act<<<cdiv((long long)N * Fin / 2, TB), TB, 0, stream>>>(x, apk, N * Fin / 2);
    k_wmma_gemm_bf16<<<(Mpad / 128) * (HC / 64), 256, 0, stream>>>(apk, w1pk, h1, N, Fin, HC);
    k_node_alpha<<<cdiv(N * H, TB), TB, 0, stream>>>(h1, a_src1, a_dst1, es1, ed1, N * H, H, C);
    k_gat_agg<<<cdiv((long long)N * H * 32, TB), TB, 0, stream>>>(
        h1, es1, ed1, rowptr, eidx, src, b1, hp, E, N, H, C, HC, 1);

    // ---- layer 2: h2 = hp @ W2 (WMMA + async-LDS B), logits, fused agg + bias ----
    k_pack_act<<<cdiv((long long)N * HC / 2, TB), TB, 0, stream>>>(hp, apk, N * HC / 2);
    k_wmma_gemm_bf16<<<(Mpad / 128) * (C / 64), 256, 0, stream>>>(apk, w2pk, h2, N, HC, C);
    k_node_alpha<<<cdiv(N, TB), TB, 0, stream>>>(h2, a_src2, a_dst2, es2, ed2, N, 1, C);
    k_gat_agg<<<cdiv((long long)N * 32, TB), TB, 0, stream>>>(
        h2, es2, ed2, rowptr, eidx, src, b2, ob, E, N, 1, C, C, 0);
  }
}
